// GuidedUpsampler_52295521796137
// MI455X (gfx1250) — matrix operand order repaired and checked by reading, not compiled, not run on hardware
//
#include <hip/hip_runtime.h>
#include <hip/hip_bf16.h>

typedef __attribute__((ext_vector_type(16))) _Float16 v16h_t;
typedef __attribute__((ext_vector_type(8)))  _Float16 v8h_t;
typedef __attribute__((ext_vector_type(8)))  float    v8f_t;

// Problem constants
#define BB   2
#define CC   128
#define HH   128
#define WW   128
#define H2   256
#define W2   256
#define EE   32
#define PADW 258          // 256 + 1 halo each side
#define NPIX (BB*H2*W2)   // 131072 output pixels (both batches)

// ---------------------------------------------------------------------------
// utility kernels
// ---------------------------------------------------------------------------
__global__ void zero_u32_kernel(unsigned int* __restrict__ p, int n) {
    int i = blockIdx.x * 256 + threadIdx.x;
    if (i < n) p[i] = 0u;
}

// (M, C, 3, 3) f32 -> f16 [m][tap][c]   (tap = dy*3+dx)
__global__ void cvt_w3x3_kernel(const float* __restrict__ src, _Float16* __restrict__ dst,
                                int C, int total) {
    int idx = blockIdx.x * 256 + threadIdx.x;
    if (idx >= total) return;
    int m   = idx / (C * 9);
    int r   = idx - m * (C * 9);
    int c   = r / 9;
    int tap = r - c * 9;
    dst[(size_t)m * 9 * C + tap * C + c] = (_Float16)src[idx];
}

__global__ void cvt_f32_to_f16_kernel(const float* __restrict__ src, _Float16* __restrict__ dst, int n) {
    int i = blockIdx.x * 256 + threadIdx.x;
    if (i < n) dst[i] = (_Float16)src[i];
}

// feat (B,C,H,W) f32 NCHW -> feat16 (B,H,W,C) f16 pixel-major
__global__ void cvt_feat_kernel(const float* __restrict__ feat, _Float16* __restrict__ feat16) {
    int idx = blockIdx.x * 256 + threadIdx.x;           // B*H*W*C = 4194304
    int c = idx & (CC - 1);
    int pix = idx >> 7;
    int x = pix & (WW - 1);
    int y = (pix >> 7) & (HH - 1);
    int b = pix >> 14;
    feat16[idx] = (_Float16)feat[(((size_t)b * CC + c) * HH + y) * WW + x];
}

// ---------------------------------------------------------------------------
// guide conv1: 3->32, 3x3, scalar (K=27, too small for WMMA). out NHWC f32.
// ---------------------------------------------------------------------------
__global__ void conv1_kernel(const float* __restrict__ guide, const float* __restrict__ w1,
                             float* __restrict__ dst) {
    int idx = blockIdx.x * 256 + threadIdx.x;           // NPIX*32
    int e = idx & 31;
    int pix = idx >> 5;
    int x = pix & (W2 - 1);
    int y = (pix >> 8) & (H2 - 1);
    int b = pix >> 16;
    float acc = 0.f;
#pragma unroll
    for (int g = 0; g < 3; ++g)
#pragma unroll
        for (int dy = 0; dy < 3; ++dy) {
            int yy = y + dy - 1;
            if (yy < 0 || yy >= H2) continue;
#pragma unroll
            for (int dx = 0; dx < 3; ++dx) {
                int xx = x + dx - 1;
                if (xx < 0 || xx >= W2) continue;
                acc += guide[(((size_t)b * 3 + g) * H2 + yy) * W2 + xx]
                     * w1[((e * 3 + g) * 3 + dy) * 3 + dx];
            }
        }
    dst[idx] = acc;
}

// ---------------------------------------------------------------------------
// deterministic BN statistics: per-block partials, then 1-block reduce
// src: (NPIX, 32) NHWC f32
// ---------------------------------------------------------------------------
__global__ void stats_partial_kernel(const float* __restrict__ src, float* __restrict__ partial) {
    int t = threadIdx.x;
    int c = t & 31, slot = t >> 5;                      // 8 slots x 32 channels
    float s = 0.f, q = 0.f;
    for (int p = blockIdx.x * 8 + slot; p < NPIX; p += 64 * 8) {
        float v = src[(size_t)p * 32 + c];
        s += v; q += v * v;
    }
    __shared__ float sh[2][8][32];
    sh[0][slot][c] = s; sh[1][slot][c] = q;
    __syncthreads();
    if (slot == 0) {
        float S = 0.f, Q = 0.f;
#pragma unroll
        for (int i = 0; i < 8; ++i) { S += sh[0][i][c]; Q += sh[1][i][c]; }
        partial[blockIdx.x * 64 + c]      = S;
        partial[blockIdx.x * 64 + 32 + c] = Q;
    }
}

__global__ void stats_reduce_kernel(const float* __restrict__ partial, float* __restrict__ stats) {
    int t = threadIdx.x;                                // 64 threads: [sum(32) | sumsq(32)]
    float s = 0.f;
    for (int b = 0; b < 64; ++b) s += partial[b * 64 + t];
    stats[t] = s;
}

// ---------------------------------------------------------------------------
// BN(train) + ReLU + f32->f16, write into padded NHWC buffer (258x258, dstC ch)
// ---------------------------------------------------------------------------
__global__ void bnrelu_kernel(const float* __restrict__ src, const float* __restrict__ stats,
                              const float* __restrict__ gamma, const float* __restrict__ beta,
                              _Float16* __restrict__ dst, int dstC, int cOff) {
    int idx = blockIdx.x * 256 + threadIdx.x;           // NPIX*32
    int c = idx & 31;
    int pix = idx >> 5;
    int x = pix & (W2 - 1);
    int y = (pix >> 8) & (H2 - 1);
    int b = pix >> 16;
    const float invN = 1.f / (float)NPIX;
    float mean = stats[c] * invN;
    float var  = stats[32 + c] * invN - mean * mean;
    float sc   = gamma[c] * rsqrtf(var + 1e-5f);
    float v    = (src[idx] - mean) * sc + beta[c];
    v = fmaxf(v, 0.f);
    dst[((((size_t)b * PADW) + y + 1) * PADW + x + 1) * dstC + cOff + c] = (_Float16)v;
}

// ---------------------------------------------------------------------------
// implicit-GEMM 3x3 conv via WMMA.  in: padded NHWC f16 (B,258,258,CIN),
// w: f16 [32][9*CIN] (k = tap*CIN + c), out: NHWC f32 (B,256,256,32).
// Block = 256 thr = 8 waves: 2 M-tiles x 4 N-tiles (64 pixels of one row).
// ---------------------------------------------------------------------------
template <int CIN, bool RELU>
__global__ void __launch_bounds__(256)
conv3x3_wmma_kernel(const _Float16* __restrict__ in, const _Float16* __restrict__ w,
                    float* __restrict__ out) {
    constexpr int K  = 9 * CIN;
    constexpr int XW = 66;                              // 64 pixels + 2 halo
    __shared__ __align__(16) _Float16 s_act[3 * XW * CIN];
    __shared__ __align__(16) _Float16 s_w[32 * K];

    const int xt = blockIdx.x * 64, y = blockIdx.y, b = blockIdx.z;
    const int tid = threadIdx.x;

    {   // stage weights (contiguous) and 3 activation rows (each 66*CIN contiguous)
        const uint4* wsrc = (const uint4*)w;
        uint4* wdst = (uint4*)s_w;
        for (int i = tid; i < 32 * K / 8; i += 256) wdst[i] = wsrc[i];
#pragma unroll
        for (int r = 0; r < 3; ++r) {
            const uint4* src = (const uint4*)(in + ((((size_t)b * PADW) + y + r) * PADW + xt) * CIN);
            uint4* dst = (uint4*)(s_act + r * XW * CIN);
            for (int i = tid; i < XW * CIN / 8; i += 256) dst[i] = src[i];
        }
    }
    __syncthreads();

    const int wave = tid >> 5, lane = tid & 31;
    const int mt = wave & 1, nt = wave >> 1;
    const bool hi = lane >= 16;
    const int l15 = lane & 15;
    const int m = mt * 16 + l15;

    v8f_t acc = {};
#pragma unroll
    for (int kk = 0; kk < K / 32; ++kk) {
        const int k0 = kk * 32;
        // A fragment: lanes 0-15 hold K k0..k0+15, lanes 16-31 hold k0+16..k0+31
        v16h_t av;
        {
            const _Float16* p = s_w + (size_t)m * K + k0 + (hi ? 16 : 0);
            v8h_t a0 = *(const v8h_t*)p;
            v8h_t a1 = *(const v8h_t*)(p + 8);
#pragma unroll
            for (int i = 0; i < 8; ++i) { av[i] = a0[i]; av[8 + i] = a1[i]; }
        }
        // B fragment: lanes 0-15 hold K k0..k0+15, lanes 16-31 hold k0+16..k0+31
        v16h_t bv;
        {
            const int kb  = k0 + (hi ? 16 : 0);
            const int tap = kb / CIN;
            const int c   = kb - tap * CIN;
            const int dy  = tap / 3, dx = tap - dy * 3;
            const int xl  = nt * 16 + l15 + dx;
            const _Float16* p = s_act + ((size_t)(dy * XW + xl)) * CIN + c;
            v8h_t b0 = *(const v8h_t*)p;
            v8h_t b1 = *(const v8h_t*)(p + 8);
#pragma unroll
            for (int i = 0; i < 8; ++i) { bv[i] = b0[i]; bv[8 + i] = b1[i]; }
        }
#if defined(__HIP_DEVICE_COMPILE__)
        acc = __builtin_amdgcn_wmma_f32_16x16x32_f16(false, av, false, bv,
                                                     (short)0, acc, false, false);
#endif
    }
    // store: lane covers 8 consecutive output channels of one pixel
    const int xp = xt + nt * 16 + l15;
    float* o = out + ((((size_t)b * H2) + y) * W2 + xp) * 32 + mt * 16 + (hi ? 8 : 0);
    v8f_t r = acc;
    if (RELU) {
#pragma unroll
        for (int i = 0; i < 8; ++i) r[i] = fmaxf(r[i], 0.f);
    }
    ((float4*)o)[0] = *(const float4*)&r;
    ((float4*)o)[1] = *((const float4*)&r + 1);
}

// ---------------------------------------------------------------------------
// projection 1x1 conv (128->32) at HALF resolution via WMMA, ReLU, replicate
// 2x2 into comb channels 32..63 (padded NHWC, 64 ch).
// ---------------------------------------------------------------------------
__global__ void __launch_bounds__(256)
proj1x1_wmma_kernel(const _Float16* __restrict__ feat16,   // (B*H*W, 128)
                    const _Float16* __restrict__ wp,       // (32, 128)
                    _Float16* __restrict__ comb) {
    __shared__ __align__(16) _Float16 s_w[32 * 128];
    const int tid = threadIdx.x;
    {
        const uint4* src = (const uint4*)wp;
        uint4* dst = (uint4*)s_w;
        for (int i = tid; i < 32 * 128 / 8; i += 256) dst[i] = src[i];
    }
    __syncthreads();

    const int wave = tid >> 5, lane = tid & 31;
    const int mt = wave & 1, nt = wave >> 1;
    const bool hi = lane >= 16;
    const int l15 = lane & 15;
    const int m = mt * 16 + l15;

    const int p = blockIdx.x * 64 + nt * 16 + l15;      // half-res pixel id
    const int b  = p >> 14;
    const int rm = p & 16383;
    const int yh = rm >> 7;
    const int xh = rm & 127;

    v8f_t acc = {};
#pragma unroll
    for (int kk = 0; kk < 4; ++kk) {                    // K = 128
        const int k0 = kk * 32;
        v16h_t av;
        {
            const _Float16* q = s_w + (size_t)m * 128 + k0 + (hi ? 16 : 0);
            v8h_t a0 = *(const v8h_t*)q;
            v8h_t a1 = *(const v8h_t*)(q + 8);
#pragma unroll
            for (int i = 0; i < 8; ++i) { av[i] = a0[i]; av[8 + i] = a1[i]; }
        }
        v16h_t bv;
        {
            const int kb = k0 + (hi ? 16 : 0);
            const _Float16* q = feat16 + (size_t)p * 128 + kb;
            v8h_t b0 = *(const v8h_t*)q;
            v8h_t b1 = *(const v8h_t*)(q + 8);
#pragma unroll
            for (int i = 0; i < 8; ++i) { bv[i] = b0[i]; bv[8 + i] = b1[i]; }
        }
#if defined(__HIP_DEVICE_COMPILE__)
        acc = __builtin_amdgcn_wmma_f32_16x16x32_f16(false, av, false, bv,
                                                     (short)0, acc, false, false);
#endif
    }
    v8h_t hv;
#pragma unroll
    for (int i = 0; i < 8; ++i) hv[i] = (_Float16)fmaxf(acc[i], 0.f);
    const int cbase = 32 + mt * 16 + (hi ? 8 : 0);
#pragma unroll
    for (int sy = 0; sy < 2; ++sy)
#pragma unroll
        for (int sx = 0; sx < 2; ++sx) {
            size_t idx = ((((size_t)b * PADW) + 2 * yh + sy + 1) * PADW + 2 * xh + sx + 1) * 64 + cbase;
            *(v8h_t*)(comb + idx) = hv;
        }
}

// ---------------------------------------------------------------------------
// 1x1 (32->9) + bias + softmax over 9.  r3 already ReLU'd, NHWC (NPIX,32).
// ---------------------------------------------------------------------------
__global__ void ksoftmax_kernel(const float* __restrict__ r3, const float* __restrict__ w2,
                                const float* __restrict__ b2, float* __restrict__ kern) {
    __shared__ float s_w[9 * 32];
    __shared__ float s_b[9];
    int t = threadIdx.x;
    if (t < 288) s_w[t] = w2[t];
    if (t < 9)   s_b[t] = b2[t];
    __syncthreads();
    int p = blockIdx.x * 256 + t;
    float x[32];
    const float4* src = (const float4*)(r3 + (size_t)p * 32);
#pragma unroll
    for (int i = 0; i < 8; ++i) {
        float4 v = src[i];
        x[4 * i] = v.x; x[4 * i + 1] = v.y; x[4 * i + 2] = v.z; x[4 * i + 3] = v.w;
    }
    float v[9], mx = -3.0e38f;
#pragma unroll
    for (int j = 0; j < 9; ++j) {
        float a = s_b[j];
#pragma unroll
        for (int c = 0; c < 32; ++c) a += s_w[j * 32 + c] * x[c];
        v[j] = a;
        mx = fmaxf(mx, a);
    }
    float sum = 0.f;
#pragma unroll
    for (int j = 0; j < 9; ++j) { v[j] = __expf(v[j] - mx); sum += v[j]; }
    float inv = 1.f / sum;
#pragma unroll
    for (int j = 0; j < 9; ++j) kern[(size_t)p * 9 + j] = v[j] * inv;
}

// ---------------------------------------------------------------------------
// CARAFE reassembly: out[b,c,y,x] = sum_i kern[i,y,x] * feat_up_pad[c,y+dy,x+dx]
// feat_up implicit from half-res feat (nearest).  Block: one (b,y,64-x tile).
// ---------------------------------------------------------------------------
__global__ void __launch_bounds__(256)
reassemble_kernel(const float* __restrict__ kern,        // (NPIX, 9)
                  const float* __restrict__ feat,        // (B,C,H,W) f32 NCHW
                  float* __restrict__ out) {             // (B,C,H2,W2)
    __shared__ float s_k[64][10];
    const int xt = blockIdx.x * 64, y = blockIdx.y, b = blockIdx.z;
    const int tid = threadIdx.x;
    for (int i = tid; i < 576; i += 256) {
        int x = i / 9, j = i - (i / 9) * 9;
        s_k[x][j] = kern[((((size_t)b * H2) + y) * W2 + xt + x) * 9 + j];
    }
    __syncthreads();

    const int x = tid & 63;
    const int c0 = tid >> 6;                             // 0..3
    const int xg = xt + x;
    for (int c = c0; c < CC; c += 4) {
        const float* fb = feat + ((size_t)b * CC + c) * HH * WW;
        __builtin_prefetch(fb + (size_t)(y >> 1) * WW + (xg >> 1), 0, 1);
        float acc = 0.f;
#pragma unroll
        for (int dy = 0; dy < 3; ++dy) {
            int yy = y + dy - 1;
            if (yy < 0 || yy >= H2) continue;
            int ry = yy >> 1;
#pragma unroll
            for (int dx = 0; dx < 3; ++dx) {
                int xx = xg + dx - 1;
                if (xx < 0 || xx >= W2) continue;
                acc += s_k[x][dy * 3 + dx] * fb[(size_t)ry * WW + (xx >> 1)];
            }
        }
        out[(((size_t)b * CC + c) * H2 + y) * W2 + xg] = acc;
    }
}

// ---------------------------------------------------------------------------
// launcher
// ---------------------------------------------------------------------------
extern "C" void kernel_launch(void* const* d_in, const int* in_sizes, int n_in,
                              void* d_out, int out_size, void* d_ws, size_t ws_size,
                              hipStream_t stream) {
    const float* feat  = (const float*)d_in[0];
    const float* guide = (const float*)d_in[1];
    const float* g_w1  = (const float*)d_in[2];
    const float* g_g1  = (const float*)d_in[3];
    const float* g_b1  = (const float*)d_in[4];
    const float* g_w2  = (const float*)d_in[5];
    const float* g_g2  = (const float*)d_in[6];
    const float* g_b2  = (const float*)d_in[7];
    const float* p_w   = (const float*)d_in[8];
    const float* k_w1  = (const float*)d_in[9];
    const float* k_w2  = (const float*)d_in[10];
    const float* k_b2  = (const float*)d_in[11];
    float* out = (float*)d_out;

    char* ws = (char*)d_ws;
    size_t off = 0;
    auto carve = [&](size_t bytes) -> void* {
        off = (off + 255) & ~(size_t)255;
        void* p = ws + off;
        off += bytes;
        return p;
    };
    _Float16* comb   = (_Float16*)carve((size_t)BB * PADW * PADW * 64 * 2);   // g|f, padded NHWC
    _Float16* g1pad  = (_Float16*)carve((size_t)BB * PADW * PADW * 32 * 2);   // stage-1 guide, padded
    float*    rA     = (float*)carve((size_t)NPIX * 32 * 4);                  // conv1 raw, later kconv raw
    float*    rB     = (float*)carve((size_t)NPIX * 32 * 4);                  // conv2 raw, later kern(9)
    _Float16* feat16 = (_Float16*)carve((size_t)BB * HH * WW * CC * 2);       // pixel-major f16
    _Float16* w2h    = (_Float16*)carve(32 * 288 * 2);
    _Float16* wkh    = (_Float16*)carve(32 * 576 * 2);
    _Float16* wph    = (_Float16*)carve(32 * 128 * 2);
    float*    part   = (float*)carve(64 * 64 * 4);
    float*    s1     = (float*)carve(64 * 4);
    float*    s2     = (float*)carve(64 * 4);
    float*    kern   = rB;                                                     // reuse after bnrelu2

    // 1. zero padded f16 buffers (borders must be 0)
    {
        int n1 = (int)((size_t)BB * PADW * PADW * 64 * 2 / 4);
        zero_u32_kernel<<<(n1 + 255) / 256, 256, 0, stream>>>((unsigned int*)comb, n1);
        int n2 = (int)((size_t)BB * PADW * PADW * 32 * 2 / 4);
        zero_u32_kernel<<<(n2 + 255) / 256, 256, 0, stream>>>((unsigned int*)g1pad, n2);
    }
    // 2. weight / feature format conversions
    cvt_w3x3_kernel<<<(32 * 32 * 9 + 255) / 256, 256, 0, stream>>>(g_w2, w2h, 32, 32 * 32 * 9);
    cvt_w3x3_kernel<<<(32 * 64 * 9 + 255) / 256, 256, 0, stream>>>(k_w1, wkh, 64, 32 * 64 * 9);
    cvt_f32_to_f16_kernel<<<(32 * 128 + 255) / 256, 256, 0, stream>>>(p_w, wph, 32 * 128);
    cvt_feat_kernel<<<(BB * HH * WW * CC) / 256, 256, 0, stream>>>(feat, feat16);

    const int nElem = NPIX * 32;                        // 4194304
    // 3-5. guide stage 1: conv + BN stats + BN/ReLU into padded f16
    conv1_kernel<<<nElem / 256, 256, 0, stream>>>(guide, g_w1, rA);
    stats_partial_kernel<<<64, 256, 0, stream>>>(rA, part);
    stats_reduce_kernel<<<1, 64, 0, stream>>>(part, s1);
    bnrelu_kernel<<<nElem / 256, 256, 0, stream>>>(rA, s1, g_g1, g_b1, g1pad, 32, 0);

    // 6-8. guide stage 2: WMMA conv 32->32 + BN stats + BN/ReLU into comb[0..31]
    conv3x3_wmma_kernel<32, false><<<dim3(4, H2, BB), 256, 0, stream>>>(g1pad, w2h, rB);
    stats_partial_kernel<<<64, 256, 0, stream>>>(rB, part);
    stats_reduce_kernel<<<1, 64, 0, stream>>>(part, s2);
    bnrelu_kernel<<<nElem / 256, 256, 0, stream>>>(rB, s2, g_g2, g_b2, comb, 64, 0);

    // 9. projection 128->32 at half-res, ReLU, replicate 2x2 into comb[32..63]
    proj1x1_wmma_kernel<<<(BB * HH * WW) / 64, 256, 0, stream>>>(feat16, wph, comb);

    // 10. kernel predictor conv 64->32 (K=576) via WMMA, ReLU fused
    conv3x3_wmma_kernel<64, true><<<dim3(4, H2, BB), 256, 0, stream>>>(comb, wkh, rA);

    // 11. 1x1 32->9 + bias + softmax
    ksoftmax_kernel<<<NPIX / 256, 256, 0, stream>>>(rA, k_w2, k_b2, kern);

    // 12. CARAFE reassembly -> d_out
    reassemble_kernel<<<dim3(4, H2, BB), 256, 0, stream>>>(kern, feat, out);
}
